// NanoGPT_15788299780423
// MI455X (gfx1250) — matrix operand order repaired and checked by reading, not compiled, not run on hardware
//
#include <hip/hip_runtime.h>

// ---------------- problem constants ----------------
enum {
  Bn = 4, Tn = 1024, En = 768, Hn = 12, HSn = 64, Ln = 6,
  FFn = 3072, Vn = 50257, VPAD = 50304, BT = Bn * Tn
};

typedef __attribute__((ext_vector_type(16))) __bf16 v16bf;
typedef __attribute__((ext_vector_type(8)))  __bf16 v8bf;
typedef __attribute__((ext_vector_type(8)))  float  v8f;

__device__ __forceinline__ v16bf pack16(v8bf lo, v8bf hi) {
  v16bf r;
#pragma unroll
  for (int j = 0; j < 8; ++j) { r[j] = lo[j]; r[j + 8] = hi[j]; }
  return r;
}

// 16-byte global -> LDS async copy (gfx1250 GLOBAL_LOAD_ASYNC_TO_LDS_B128, GVS mode)
__device__ __forceinline__ void async_cp16(unsigned lds_off, const __bf16* base,
                                           unsigned gbyte_off) {
  asm volatile("global_load_async_to_lds_b128 %0, %1, %2"
               :: "v"(lds_off), "v"(gbyte_off), "s"(base) : "memory");
}

// ---------------- weight conversion (fp32 -> bf16, pre-transposed to [N][K]) --------
// Wq/Wk/Wv: [L][H][E][HS] -> [L][H*HS][E]   (B^T layout: row n = h*64+d, col k = e)
__global__ void k_cvt_head(const float* __restrict__ w, __bf16* __restrict__ o, int n) {
  int i = blockIdx.x * 256 + threadIdx.x;
  if (i >= n) return;
  int d = i & 63;
  int t = i >> 6;
  int e = t % En; t /= En;
  int h = t % Hn;
  int l = t / Hn;
  o[((size_t)l * En + h * HSn + d) * En + e] = (__bf16)w[i];
}

// generic per-layer transpose: in [L][Kd][Nd] -> out [L][Nd][Kd]
__global__ void k_cvt_T(const float* __restrict__ w, __bf16* __restrict__ o,
                        int Kd, int Nd, int n) {
  int i = blockIdx.x * 256 + threadIdx.x;
  if (i >= n) return;
  int kn = Kd * Nd;
  int l = i / kn, rem = i % kn;
  int k = rem / Nd, nn = rem % Nd;
  o[(size_t)l * kn + (size_t)nn * Kd + k] = (__bf16)w[i];
}

// Wlm: [E][V] -> [VPAD][E] bf16 (zero padded rows)
__global__ void k_cvt_lmT(const float* __restrict__ w, __bf16* __restrict__ o) {
  int i = blockIdx.x * 256 + threadIdx.x;
  if (i >= VPAD * En) return;
  int n = i / En, k = i % En;
  o[i] = (n < Vn) ? (__bf16)w[(size_t)k * Vn + n] : (__bf16)0.0f;
}

// ---------------- embedding ----------------
__global__ __launch_bounds__(256) void k_embed(const int* __restrict__ idx,
                                               const float* __restrict__ tok,
                                               const float* __restrict__ pos,
                                               float* __restrict__ x) {
  int row = blockIdx.x, tid = threadIdx.x;
  int t = row & (Tn - 1);
  int id = idx[row];
#pragma unroll
  for (int i = 0; i < 3; ++i) {
    int c = tid + i * 256;
    x[(size_t)row * En + c] = tok[(size_t)id * En + c] + pos[(size_t)t * En + c];
  }
}

// ---------------- layernorm (fp32 in -> bf16 out) ----------------
__global__ __launch_bounds__(256) void k_ln(const float* __restrict__ x,
                                            const float* __restrict__ g,
                                            const float* __restrict__ bta,
                                            __bf16* __restrict__ out) {
  __shared__ float red[256];
  int row = blockIdx.x, tid = threadIdx.x;
  const float* xr = x + (size_t)row * En;
  float v0 = xr[tid], v1 = xr[tid + 256], v2 = xr[tid + 512];
  red[tid] = v0 + v1 + v2;
  __syncthreads();
  for (int st = 128; st > 0; st >>= 1) {
    if (tid < st) red[tid] += red[tid + st];
    __syncthreads();
  }
  float mu = red[0] * (1.0f / En);
  __syncthreads();
  float d0 = v0 - mu, d1 = v1 - mu, d2 = v2 - mu;
  red[tid] = d0 * d0 + d1 * d1 + d2 * d2;
  __syncthreads();
  for (int st = 128; st > 0; st >>= 1) {
    if (tid < st) red[tid] += red[tid + st];
    __syncthreads();
  }
  float rs = rsqrtf(red[0] * (1.0f / En) + 1e-5f);
  __bf16* orow = out + (size_t)row * En;
  orow[tid]       = (__bf16)(d0 * rs * g[tid]       + bta[tid]);
  orow[tid + 256] = (__bf16)(d1 * rs * g[tid + 256] + bta[tid + 256]);
  orow[tid + 512] = (__bf16)(d2 * rs * g[tid + 512] + bta[tid + 512]);
}

// ---------------- WMMA GEMM: C[M,N] = A[M,K](bf16,row-major) * Bt[N,K](bf16)^T ------
// WG 256 thr = 8 waves; WG tile 128x128; wave tile 32x64 (2x4 wmma accumulators).
// Double-buffered async global->LDS staging, 32 K per stage.
__global__ __launch_bounds__(256) void k_gemm(
    const __bf16* __restrict__ A, const __bf16* __restrict__ Bt,
    const float* __restrict__ bias, const float* __restrict__ resid,
    float* __restrict__ outF, __bf16* __restrict__ outB, __bf16* __restrict__ outVT,
    int M, int N, int K, int Nvalid, int ldC, int relu) {
  __shared__ __bf16 As[2][128 * 40];  // [row][k0..31], row stride 40 elems (80 B)
  __shared__ __bf16 Bs[2][128 * 40];  // [col][k0..31], col stride 40 elems
  const int tid = threadIdx.x;
  const int lane = tid & 31, wave = tid >> 5;
  const int hs = lane >> 4, lr = lane & 15;
  const int m0 = blockIdx.y * 128, n0 = blockIdx.x * 128;
  const int wm = (wave >> 1) * 32;   // 0,32,64,96
  const int wn = (wave & 1) * 64;    // 0,64

  // per-thread staging geometry: 2 A-chunks + 2 B-chunks of 16 B per stage
  const int r0 = tid >> 2, ch = tid & 3;           // rows r0, r0+64 ; 16B chunk ch
  const unsigned asb = (unsigned)(uintptr_t)&As[0][0];
  const unsigned bsb = (unsigned)(uintptr_t)&Bs[0][0];
  const unsigned ldsA0 = asb + (unsigned)(r0 * 80 + ch * 16);
  const unsigned ldsA1 = asb + (unsigned)((r0 + 64) * 80 + ch * 16);
  const unsigned ldsB0 = bsb + (unsigned)(r0 * 80 + ch * 16);
  const unsigned ldsB1 = bsb + (unsigned)((r0 + 64) * 80 + ch * 16);
  const unsigned stgOff = 128 * 40 * 2;            // bytes between stage buffers
  const unsigned gA0 = (unsigned)(((size_t)(m0 + r0) * K + ch * 8) * 2);
  const unsigned gA1 = (unsigned)(((size_t)(m0 + r0 + 64) * K + ch * 8) * 2);
  const unsigned gB0 = (unsigned)(((size_t)(n0 + r0) * K + ch * 8) * 2);
  const unsigned gB1 = (unsigned)(((size_t)(n0 + r0 + 64) * K + ch * 8) * 2);

  v8f acc[2][4] = {};

  // prologue: stage 0
  async_cp16(ldsA0, A, gA0);
  async_cp16(ldsA1, A, gA1);
  async_cp16(ldsB0, Bt, gB0);
  async_cp16(ldsB1, Bt, gB1);

  int stage = 0;
  for (int k0 = 0; k0 < K; k0 += 32, stage ^= 1) {
    const int kn = k0 + 32;
    if (kn < K) {
      const unsigned so = (stage ^ 1) ? stgOff : 0u;
      const unsigned kb = (unsigned)(kn * 2);
      async_cp16(ldsA0 + so, A, gA0 + kb);
      async_cp16(ldsA1 + so, A, gA1 + kb);
      async_cp16(ldsB0 + so, Bt, gB0 + kb);
      async_cp16(ldsB1 + so, Bt, gB1 + kb);
      asm volatile("s_wait_asynccnt 0x4" ::: "memory");
    } else {
      asm volatile("s_wait_asynccnt 0x0" ::: "memory");
    }
    __syncthreads();

    v16bf af[2], bf[4];
#pragma unroll
    for (int mi = 0; mi < 2; ++mi) {
      // A 16x32 frag: lanes<16 -> K0..7 & K16..23 ; lanes>=16 -> K8..15 & K24..31
      const __bf16* p = &As[stage][(wm + mi * 16 + lr) * 40 + hs * 8];
      af[mi] = pack16(*(const v8bf*)p, *(const v8bf*)(p + 16));
    }
#pragma unroll
    for (int ni = 0; ni < 4; ++ni) {
      // B 32x16 frag: col = lane%16 ; lanes<16 -> K0..7 & K16..23 ; lanes>=16 -> K8..15 & K24..31
      const __bf16* p = &Bs[stage][(wn + ni * 16 + lr) * 40 + hs * 8];
      bf[ni] = pack16(*(const v8bf*)p, *(const v8bf*)(p + 16));
    }
#pragma unroll
    for (int mi = 0; mi < 2; ++mi)
#pragma unroll
      for (int ni = 0; ni < 4; ++ni)
        acc[mi][ni] = __builtin_amdgcn_wmma_f32_16x16x32_bf16(
            false, af[mi], false, bf[ni], (short)0, acc[mi][ni], false, false);
    __syncthreads();
  }

  // epilogue: C layout = reg r: lanes<16 row r, lanes>=16 row r+8; col = lane%16
#pragma unroll
  for (int mi = 0; mi < 2; ++mi)
#pragma unroll
    for (int ni = 0; ni < 4; ++ni) {
      int gn = n0 + wn + ni * 16 + lr;
      if (gn >= Nvalid) continue;
      float bv = bias ? bias[gn] : 0.0f;
#pragma unroll
      for (int r = 0; r < 8; ++r) {
        int gm = m0 + wm + mi * 16 + r + hs * 8;
        float v = acc[mi][ni][r] + bv;
        if (relu) v = v > 0.0f ? v : 0.0f;
        if (resid) v += resid[(size_t)gm * ldC + gn];
        if (outF) outF[(size_t)gm * ldC + gn] = v;
        if (outB) outB[(size_t)gm * ldC + gn] = (__bf16)v;
        if (outVT) {  // V transposed per (b,h): vT[((b*H+h)*64+d)][t]
          int bI = gm >> 10, tI = gm & (Tn - 1);
          int hI = gn >> 6, dI = gn & 63;
          outVT[(((size_t)(bI * Hn + hI) * HSn + dI) << 10) + tI] = (__bf16)v;
        }
      }
    }
}

// ---------------- flash attention (causal), bf16 WMMA ----------------
// grid: (T/64, B*H); block 128 = 4 waves, each wave owns 16 query rows.
// q,k row-major [BT][E]; v pre-transposed vT[(b*H+h)*64+d][T].
__global__ __launch_bounds__(128) void k_attn(const __bf16* __restrict__ qb,
                                              const __bf16* __restrict__ kb,
                                              const __bf16* __restrict__ vT,
                                              __bf16* __restrict__ ob) {
  __shared__ __bf16 P[4][16 * 40];  // per-wave P transpose scratch
  const int lane = threadIdx.x & 31, wave = threadIdx.x >> 5;
  const int hs = lane >> 4, lr = lane & 15;
  const int bh = blockIdx.y, b = bh / Hn, h = bh % Hn;
  const int t0 = blockIdx.x * 64 + wave * 16;
  const size_t base = (size_t)b * Tn * En + h * HSn;
  const size_t vbase = (size_t)bh * HSn * Tn;
  const float scale = 0.125f;  // 1/sqrt(64)

  v16bf qa[2];
#pragma unroll
  for (int dh = 0; dh < 2; ++dh) {
    const __bf16* p = qb + base + (size_t)(t0 + lr) * En + dh * 32 + hs * 8;
    qa[dh] = pack16(*(const v8bf*)p, *(const v8bf*)(p + 16));
  }

  v8f oacc[4] = {};
  float mrow[8], lrow[8];
#pragma unroll
  for (int r = 0; r < 8; ++r) { mrow[r] = -1e30f; lrow[r] = 0.0f; }

  const int nch = (t0 + 16 + 31) >> 5;  // 32-key chunks covering keys 0..t0+15
  for (int c = 0; c < nch; ++c) {
    const int s0 = c * 32;
    v8f S[2] = {};
#pragma unroll
    for (int kt = 0; kt < 2; ++kt)
#pragma unroll
      for (int dh = 0; dh < 2; ++dh) {
        const __bf16* p = kb + base + (size_t)(s0 + kt * 16 + lr) * En + dh * 32 + hs * 8;
        v16bf bfr = pack16(*(const v8bf*)p, *(const v8bf*)(p + 16));
        S[kt] = __builtin_amdgcn_wmma_f32_16x16x32_bf16(
            false, qa[dh], false, bfr, (short)0, S[kt], false, false);
      }

    float mc[8];
#pragma unroll
    for (int r = 0; r < 8; ++r) {
      int row = t0 + r + hs * 8;
#pragma unroll
      for (int kt = 0; kt < 2; ++kt) {
        int key = s0 + kt * 16 + lr;
        float sv = S[kt][r] * scale;
        if (key > row) sv = -1e30f;
        S[kt][r] = sv;
      }
      float m2 = S[0][r] > S[1][r] ? S[0][r] : S[1][r];
#pragma unroll
      for (int msk = 1; msk < 16; msk <<= 1) {
        float o = __shfl_xor(m2, msk, 32);
        m2 = o > m2 ? o : m2;
      }
      mc[r] = m2;
    }

#pragma unroll
    for (int r = 0; r < 8; ++r) {
      float mn = mrow[r] > mc[r] ? mrow[r] : mc[r];
      float al = expf(mrow[r] - mn);
      float ps = 0.0f;
#pragma unroll
      for (int kt = 0; kt < 2; ++kt) {
        float pv = expf(S[kt][r] - mn);
        ps += pv;
        P[wave][(r + hs * 8) * 40 + kt * 16 + lr] = (__bf16)pv;
      }
#pragma unroll
      for (int msk = 1; msk < 16; msk <<= 1) ps += __shfl_xor(ps, msk, 32);
      lrow[r] = lrow[r] * al + ps;
      mrow[r] = mn;
#pragma unroll
      for (int nt = 0; nt < 4; ++nt) oacc[nt][r] *= al;
    }

    // DS ops are in-order per wave: wait stores, reload as A fragment
    asm volatile("s_wait_dscnt 0" ::: "memory");
    const __bf16* pp = &P[wave][lr * 40 + hs * 8];
    v16bf pa = pack16(*(const v8bf*)pp, *(const v8bf*)(pp + 16));

#pragma unroll
    for (int nt = 0; nt < 4; ++nt) {
      const __bf16* pv = vT + vbase + (size_t)(nt * 16 + lr) * Tn + s0 + hs * 8;
      v16bf vf = pack16(*(const v8bf*)pv, *(const v8bf*)(pv + 16));
      oacc[nt] = __builtin_amdgcn_wmma_f32_16x16x32_bf16(
          false, pa, false, vf, (short)0, oacc[nt], false, false);
    }
  }

#pragma unroll
  for (int r = 0; r < 8; ++r) {
    float inv = 1.0f / lrow[r];
    int row = t0 + r + hs * 8;
#pragma unroll
    for (int nt = 0; nt < 4; ++nt)
      ob[base + (size_t)row * En + nt * 16 + lr] = (__bf16)(oacc[nt][r] * inv);
  }
}

// ---------------- loss ----------------
__global__ __launch_bounds__(256) void k_rownll(const float* __restrict__ logits,
                                                const int* __restrict__ tgt,
                                                float* __restrict__ nll) {
  __shared__ float red[256];
  int row = blockIdx.x, tid = threadIdx.x;
  const float* lrp = logits + (size_t)row * Vn;
  float mx = -1e30f;
  for (int j = tid; j < Vn; j += 256) { float v = lrp[j]; mx = v > mx ? v : mx; }
  red[tid] = mx;
  __syncthreads();
  for (int st = 128; st > 0; st >>= 1) {
    if (tid < st) { float o = red[tid + st]; if (o > red[tid]) red[tid] = o; }
    __syncthreads();
  }
  mx = red[0];
  __syncthreads();
  float s = 0.0f;
  for (int j = tid; j < Vn; j += 256) s += expf(lrp[j] - mx);
  red[tid] = s;
  __syncthreads();
  for (int st = 128; st > 0; st >>= 1) {
    if (tid < st) red[tid] += red[tid + st];
    __syncthreads();
  }
  if (tid == 0) nll[row] = -(lrp[tgt[row]] - mx - logf(red[0]));
}

__global__ __launch_bounds__(256) void k_loss(const float* __restrict__ nll,
                                              float* __restrict__ out) {
  __shared__ float red[256];
  int tid = threadIdx.x;
  float s = 0.0f;
  for (int j = tid; j < BT; j += 256) s += nll[j];
  red[tid] = s;
  __syncthreads();
  for (int st = 128; st > 0; st >>= 1) {
    if (tid < st) red[tid] += red[tid + st];
    __syncthreads();
  }
  if (tid == 0) out[(size_t)BT * Vn] = red[0] / (float)BT;
}

// ---------------- host orchestration ----------------
extern "C" void kernel_launch(void* const* d_in, const int* in_sizes, int n_in,
                              void* d_out, int out_size, void* d_ws, size_t ws_size,
                              hipStream_t stream) {
  (void)in_sizes; (void)n_in; (void)out_size; (void)ws_size;
  const int*   idx   = (const int*)d_in[0];
  const int*   tgt   = (const int*)d_in[1];
  const float* tok   = (const float*)d_in[2];
  const float* pos   = (const float*)d_in[3];
  const float* Wq    = (const float*)d_in[4];
  const float* Wk    = (const float*)d_in[5];
  const float* Wv    = (const float*)d_in[6];
  const float* Wo    = (const float*)d_in[7];
  const float* bo    = (const float*)d_in[8];
  const float* ln1s  = (const float*)d_in[9];
  const float* ln1b  = (const float*)d_in[10];
  const float* ln2s  = (const float*)d_in[11];
  const float* ln2b  = (const float*)d_in[12];
  const float* W1    = (const float*)d_in[13];
  const float* b1    = (const float*)d_in[14];
  const float* W2    = (const float*)d_in[15];
  const float* b2    = (const float*)d_in[16];
  const float* lnfs  = (const float*)d_in[17];
  const float* lnfb  = (const float*)d_in[18];
  const float* Wlm   = (const float*)d_in[19];
  const float* blm   = (const float*)d_in[20];
  float* logits = (float*)d_out;

  char* ws = (char*)d_ws;
  size_t off = 0;
  auto alloc = [&](size_t bytes) -> char* {
    char* p = ws + off;
    off = (off + bytes + 255) & ~(size_t)255;
    return p;
  };
  const size_t wEE = (size_t)Ln * En * En;
  const size_t wEF = (size_t)Ln * En * FFn;
  __bf16* wq_b  = (__bf16*)alloc(wEE * 2);
  __bf16* wk_b  = (__bf16*)alloc(wEE * 2);
  __bf16* wv_b  = (__bf16*)alloc(wEE * 2);
  __bf16* wo_b  = (__bf16*)alloc(wEE * 2);
  __bf16* w1_b  = (__bf16*)alloc(wEF * 2);
  __bf16* w2_b  = (__bf16*)alloc(wEF * 2);
  __bf16* wlm_b = (__bf16*)alloc((size_t)VPAD * En * 2);
  float*  x     = (float*) alloc((size_t)BT * En * 4);
  __bf16* hb    = (__bf16*)alloc((size_t)BT * En * 2);
  __bf16* qb_   = (__bf16*)alloc((size_t)BT * En * 2);
  __bf16* kb_   = (__bf16*)alloc((size_t)BT * En * 2);
  __bf16* vT_   = (__bf16*)alloc((size_t)BT * En * 2);
  __bf16* ob_   = (__bf16*)alloc((size_t)BT * En * 2);
  __bf16* fb_   = (__bf16*)alloc((size_t)BT * FFn * 2);
  __bf16* xfb_  = (__bf16*)alloc((size_t)BT * En * 2);
  float*  nll_  = (float*) alloc((size_t)BT * 4);

  {  // weight conversion (fp32 -> bf16, transposed to [N][K])
    int nh = Ln * Hn * En * HSn;
    int gb = (nh + 255) / 256;
    k_cvt_head<<<gb, 256, 0, stream>>>(Wq, wq_b, nh);
    k_cvt_head<<<gb, 256, 0, stream>>>(Wk, wk_b, nh);
    k_cvt_head<<<gb, 256, 0, stream>>>(Wv, wv_b, nh);
    int nwo = (int)wEE;
    k_cvt_T<<<(nwo + 255) / 256, 256, 0, stream>>>(Wo, wo_b, En, En, nwo);
    int nwf = (int)wEF;
    k_cvt_T<<<(nwf + 255) / 256, 256, 0, stream>>>(W1, w1_b, En, FFn, nwf);
    k_cvt_T<<<(nwf + 255) / 256, 256, 0, stream>>>(W2, w2_b, FFn, En, nwf);
    k_cvt_lmT<<<(VPAD * En + 255) / 256, 256, 0, stream>>>(Wlm, wlm_b);
  }

  auto gemm = [&](const __bf16* A, const __bf16* Bt, const float* bias,
                  const float* resid, float* outF, __bf16* outB, __bf16* outVT,
                  int M, int N, int K, int Nvalid, int ldC, int relu) {
    dim3 grid(N / 128, M / 128);
    k_gemm<<<grid, 256, 0, stream>>>(A, Bt, bias, resid, outF, outB, outVT,
                                     M, N, K, Nvalid, ldC, relu);
  };

  k_embed<<<BT, 256, 0, stream>>>(idx, tok, pos, x);

  for (int l = 0; l < Ln; ++l) {
    const size_t oEE = (size_t)l * En * En;
    const size_t oEF = (size_t)l * En * FFn;
    k_ln<<<BT, 256, 0, stream>>>(x, ln1s + l * En, ln1b + l * En, hb);
    gemm(hb, wq_b + oEE, nullptr, nullptr, nullptr, qb_, nullptr, BT, En, En, En, En, 0);
    gemm(hb, wk_b + oEE, nullptr, nullptr, nullptr, kb_, nullptr, BT, En, En, En, En, 0);
    gemm(hb, wv_b + oEE, nullptr, nullptr, nullptr, nullptr, vT_, BT, En, En, En, En, 0);
    k_attn<<<dim3(Tn / 64, Bn * Hn), 128, 0, stream>>>(qb_, kb_, vT_, ob_);
    gemm(ob_, wo_b + oEE, bo + l * En, x, x, nullptr, nullptr, BT, En, En, En, En, 0);
    k_ln<<<BT, 256, 0, stream>>>(x, ln2s + l * En, ln2b + l * En, hb);
    gemm(hb, w1_b + oEF, b1 + l * FFn, nullptr, nullptr, fb_, nullptr, BT, FFn, En, FFn, FFn, 1);
    gemm(fb_, w2_b + oEF, b2 + l * En, x, x, nullptr, nullptr, BT, En, FFn, En, En, 0);
  }

  k_ln<<<BT, 256, 0, stream>>>(x, lnfs, lnfb, xfb_);
  gemm(xfb_, wlm_b, blm, nullptr, logits, nullptr, nullptr, BT, VPAD, En, Vn, Vn, 0);

  k_rownll<<<BT, 256, 0, stream>>>(logits, tgt, nll_);
  k_loss<<<1, 256, 0, stream>>>(nll_, logits);
}
